// MMOE_82575041233053
// MI455X (gfx1250) — compile-verified, matrix operand order checked by reading, not run on hardware
//
#include <hip/hip_runtime.h>
#include <hip/hip_bf16.h>

#define BS   32768
#define NIN  512
#define E    8
#define T    2
#define H    256

typedef __attribute__((ext_vector_type(16))) __bf16 v16bf;
typedef __attribute__((ext_vector_type(8)))  __bf16 v8bf;
typedef __attribute__((ext_vector_type(8)))  float  v8f;

__device__ __forceinline__ unsigned short f32_to_bf16(float f) {
    unsigned int u = __float_as_uint(f);
    unsigned int r = (u + 0x7FFFu + ((u >> 16) & 1u)) >> 16;
    return (unsigned short)r;
}

// Load a 16-element bf16 fragment: 8 elems at p, 8 elems at p+16 (the ISA's
// per-lane A/B layout for 16-bit WMMA: K {0..7,16..23} or {8..15,24..31}).
__device__ __forceinline__ v16bf load_frag(const unsigned short* p) {
    v8bf a = *(const v8bf*)(p);
    v8bf b = *(const v8bf*)(p + 16);
    return __builtin_shufflevector(a, b, 0,1,2,3,4,5,6,7,8,9,10,11,12,13,14,15);
}

__device__ __forceinline__ v8f wmma_bf16(v16bf a, v16bf b, v8f c) {
    return __builtin_amdgcn_wmma_f32_16x16x32_bf16(
        /*neg_a=*/false, a, /*neg_b=*/false, b,
        /*c_mod=*/(short)0, c, /*reuse_a=*/false, /*reuse_b=*/false);
}

// --- prep: We[e][k][h] (f32) -> WeT[e][h][k] (bf16) -------------------------
__global__ void mmoe_prep_WeT(const float* __restrict__ We,
                              unsigned short* __restrict__ WeT) {
    int i = blockIdx.x * blockDim.x + threadIdx.x;
    if (i >= E * H * NIN) return;
    int k = i & (NIN - 1);
    int h = (i >> 9) & (H - 1);
    int e = i >> 17;
    WeT[i] = f32_to_bf16(We[((size_t)e * NIN + k) * H + h]);
}

// --- prep: Wg[t][k][e] (f32) -> WgT[t*8+e][k] (bf16) ------------------------
__global__ void mmoe_prep_WgT(const float* __restrict__ Wg,
                              unsigned short* __restrict__ WgT) {
    int i = blockIdx.x * blockDim.x + threadIdx.x;
    if (i >= T * E * NIN) return;
    int k = i & (NIN - 1);
    int c = i >> 9;            // 0..15
    int t = c >> 3;
    int e = c & 7;
    WgT[i] = f32_to_bf16(Wg[((size_t)t * NIN + k) * E + e]);
}

// --- fused MMOE kernel ------------------------------------------------------
// block = 256 threads (8 waves), owns 32 batch rows; wave w owns H cols [32w,32w+32)
#define XPAD 520   // padded LDS row stride (bf16 elems) to spread banks

__global__ void __launch_bounds__(256, 2)
mmoe_fused(const float* __restrict__ xv,
           const unsigned short* __restrict__ WeT,
           const float* __restrict__ be,
           const unsigned short* __restrict__ WgT,
           const float* __restrict__ bg,
           const float* __restrict__ Wt,
           const float* __restrict__ bt,
           float* __restrict__ out) {
    __shared__ __align__(16) unsigned short lds_x[32 * XPAD]; // xv tile (bf16)
    __shared__ float lds_g[32 * 16];                          // logits -> gates
    __shared__ float lds_part[64];                            // tower partials

    const int tid   = threadIdx.x;
    const int wid   = tid >> 5;
    const int lane  = tid & 31;
    const int lm    = lane & 15;
    const int hw    = lane >> 4;          // half-wave select
    const int nbase = wid * 32;           // this wave's H-column base
    const int rowBase = blockIdx.x * 32;

    // 1) stage xv tile (32 x 512 f32) into LDS as bf16
    for (int i = tid; i < 32 * NIN; i += 256) {
        int r = i >> 9, k = i & (NIN - 1);
        lds_x[r * XPAD + k] = f32_to_bf16(xv[(size_t)(rowBase + r) * NIN + k]);
    }
    __syncthreads();

    // 2) gate logits via WMMA (wave 0 only): 32 rows x 16 (t,e) cols
    if (wid == 0) {
        v8f gl0 = {}, gl1 = {};
        for (int kt = 0; kt < 16; ++kt) {
            int kA = kt * 32 + hw * 8;
            v16bf a0 = load_frag(lds_x + lm * XPAD + kA);
            v16bf a1 = load_frag(lds_x + (16 + lm) * XPAD + kA);
            v16bf bgf = load_frag(WgT + lm * NIN + kA);
            gl0 = wmma_bf16(a0, bgf, gl0);
            gl1 = wmma_bf16(a1, bgf, gl1);
        }
        #pragma unroll
        for (int r = 0; r < 8; ++r) {
            lds_g[(8 * hw + r) * 16 + lm]      = gl0[r];
            lds_g[(16 + 8 * hw + r) * 16 + lm] = gl1[r];
        }
    }
    __syncthreads();

    // 3) softmax over experts (64 threads: one per (row,t))
    if (tid < 64) {
        int row = tid >> 1, t = tid & 1;
        float l[E];
        float mx = -1e30f;
        #pragma unroll
        for (int e = 0; e < E; ++e) {
            l[e] = lds_g[row * 16 + t * 8 + e] + bg[t * E + e];
            mx = fmaxf(mx, l[e]);
        }
        float s = 0.f;
        #pragma unroll
        for (int e = 0; e < E; ++e) { l[e] = __expf(l[e] - mx); s += l[e]; }
        float inv = 1.f / s;
        #pragma unroll
        for (int e = 0; e < E; ++e) lds_g[row * 16 + t * 8 + e] = l[e] * inv;
        lds_part[tid] = 0.f;
    }
    __syncthreads();

    // 4) expert GEMMs with fused bias+ReLU+gated accumulation
    v8f acc[2][2][2] = {};  // [task][mi][ni]
    #pragma unroll 1
    for (int e = 0; e < E; ++e) {
        v8f ye[2][2] = {};
        const unsigned short* Be = WeT + (size_t)e * H * NIN;
        for (int kt = 0; kt < 16; ++kt) {
            int kA = kt * 32 + hw * 8;
            v16bf a0 = load_frag(lds_x + lm * XPAD + kA);
            v16bf a1 = load_frag(lds_x + (16 + lm) * XPAD + kA);
            v16bf b0 = load_frag(Be + (size_t)(nbase + lm) * NIN + kA);
            v16bf b1 = load_frag(Be + (size_t)(nbase + 16 + lm) * NIN + kA);
            ye[0][0] = wmma_bf16(a0, b0, ye[0][0]);
            ye[0][1] = wmma_bf16(a0, b1, ye[0][1]);
            ye[1][0] = wmma_bf16(a1, b0, ye[1][0]);
            ye[1][1] = wmma_bf16(a1, b1, ye[1][1]);
        }
        float be0 = be[e * H + nbase + lm];
        float be1 = be[e * H + nbase + 16 + lm];
        #pragma unroll
        for (int mi = 0; mi < 2; ++mi)
            #pragma unroll
            for (int ni = 0; ni < 2; ++ni)
                #pragma unroll
                for (int r = 0; r < 8; ++r) {
                    int rowl = 16 * mi + 8 * hw + r;
                    float y = ye[mi][ni][r] + (ni ? be1 : be0);
                    y = fmaxf(y, 0.f);
                    float g0 = lds_g[rowl * 16 + e];       // task 0 gate
                    float g1 = lds_g[rowl * 16 + 8 + e];   // task 1 gate
                    acc[0][mi][ni][r] += g0 * y;
                    acc[1][mi][ni][r] += g1 * y;
                }
    }

    // 5) tower: dot with Wt over this wave's 32 columns, reduce across lanes
    const int c0 = nbase + lm, c1 = nbase + 16 + lm;
    float wt00 = Wt[c0],     wt01 = Wt[c1];
    float wt10 = Wt[H + c0], wt11 = Wt[H + c1];
    #pragma unroll
    for (int t = 0; t < 2; ++t)
        #pragma unroll
        for (int mi = 0; mi < 2; ++mi)
            #pragma unroll
            for (int r = 0; r < 8; ++r) {
                float v = acc[t][mi][0][r] * (t ? wt10 : wt00)
                        + acc[t][mi][1][r] * (t ? wt11 : wt01);
                v += __shfl_xor(v, 1, 32);
                v += __shfl_xor(v, 2, 32);
                v += __shfl_xor(v, 4, 32);
                v += __shfl_xor(v, 8, 32);
                if (lm == 0)
                    atomicAdd(&lds_part[(16 * mi + 8 * hw + r) * 2 + t], v);
            }
    __syncthreads();

    // 6) bias + ReLU + sigmoid, write both task outputs
    if (tid < 64) {
        int row = tid >> 1, t = tid & 1;
        float s = lds_part[tid] + bt[t];
        s = fmaxf(s, 0.f);
        out[(size_t)t * BS + rowBase + row] = 1.f / (1.f + __expf(-s));
    }
}

extern "C" void kernel_launch(void* const* d_in, const int* in_sizes, int n_in,
                              void* d_out, int out_size, void* d_ws, size_t ws_size,
                              hipStream_t stream) {
    const float* xv = (const float*)d_in[0];
    const float* We = (const float*)d_in[1];
    const float* be = (const float*)d_in[2];
    const float* Wg = (const float*)d_in[3];
    const float* bg = (const float*)d_in[4];
    const float* Wt = (const float*)d_in[5];
    const float* bt = (const float*)d_in[6];
    float* out = (float*)d_out;

    unsigned short* WeT = (unsigned short*)d_ws;                    // E*H*NIN bf16 = 2 MB
    unsigned short* WgT = WeT + (size_t)E * H * NIN;                // T*E*NIN bf16 = 16 KB

    mmoe_prep_WeT<<<(E * H * NIN + 255) / 256, 256, 0, stream>>>(We, WeT);
    mmoe_prep_WgT<<<(T * E * NIN + 255) / 256, 256, 0, stream>>>(Wg, WgT);
    mmoe_fused<<<BS / 32, 256, 0, stream>>>(xv, WeT, be, WgT, bg, Wt, bt, out);
}
